// Seq2SeqWithAttention_27487790694939
// MI455X (gfx1250) — matrix operand order repaired and checked by reading, not compiled, not run on hardware
//
#include <hip/hip_runtime.h>

// ---------- types ----------
typedef __attribute__((ext_vector_type(8)))  __bf16 bf16x8;
typedef __attribute__((ext_vector_type(16))) __bf16 bf16x16;
typedef __attribute__((ext_vector_type(8)))  float  f32x8;

#define DEVINL __device__ __forceinline__

DEVINL float sigm(float x) { return 1.0f / (1.0f + __expf(-x)); }

DEVINL f32x8 zero8() { f32x8 z = {0.f,0.f,0.f,0.f,0.f,0.f,0.f,0.f}; return z; }

// A fragment: 16x32 bf16, row-major A[M,K]. lane<16 row m holds K{0..7,16..23},
// lane>=16 holds K{8..15,24..31} (per CDNA5 ISA 16-bit A layout).
DEVINL bf16x16 load_a_frag(const __bf16* A, int lda, int mbase, int k0, int lane) {
  int m = mbase + (lane & 15);
  const __bf16* p = A + (size_t)m * lda + k0 + ((lane & 16) ? 8 : 0);
  bf16x8 lo = *(const bf16x8*)(p);
  bf16x8 hi = *(const bf16x8*)(p + 16);
  return __builtin_shufflevector(lo, hi, 0,1,2,3,4,5,6,7,8,9,10,11,12,13,14,15);
}

// B fragment: 32x16 (KxN) where B[k][n] = W[n][k], W stored row-major [N,K].
// lane<16: column n, K{0..7,16..23} ; lane>=16: column n, K{8..15,24..31}.
DEVINL bf16x16 load_b_frag(const __bf16* W, int ldw, int nbase, int k0, int lane) {
  int n = nbase + (lane & 15);
  const __bf16* p = W + (size_t)n * ldw + k0 + ((lane & 16) ? 8 : 0);
  bf16x8 lo = *(const bf16x8*)(p);
  bf16x8 hi = *(const bf16x8*)(p + 16);
  return __builtin_shufflevector(lo, hi, 0,1,2,3,4,5,6,7,8,9,10,11,12,13,14,15);
}

DEVINL f32x8 wmma_bf16(bf16x16 a, bf16x16 b, f32x8 c) {
  return __builtin_amdgcn_wmma_f32_16x16x32_bf16(false, a, false, b, (short)0, c,
                                                 false, false);
}

// per-wave tiled GEMM into an [Mt*16, Nt*16] fp32 buffer (LDS or global),
// ldc = Nt*16.  W is [N,K] row-major bf16.  8 waves per block assumed.
DEVINL void wave_gemm(const __bf16* A, const __bf16* W, const float* bias, float* Cs,
                      int Mt, int Nt, int K, int wave, int lane) {
  int total = Mt * Nt;
  int ldc = Nt * 16;
  for (int tt = wave; tt < total; tt += 8) {
    int mt = tt / Nt, nt = tt % Nt;
    f32x8 acc = zero8();
    for (int k0 = 0; k0 < K; k0 += 32)
      acc = wmma_bf16(load_a_frag(A, K, mt*16, k0, lane),
                      load_b_frag(W, K, nt*16, k0, lane), acc);
    int col = nt*16 + (lane & 15);
    int mo  = (lane & 16) ? 8 : 0;
    float bv = bias ? bias[col] : 0.f;
#pragma unroll
    for (int i = 0; i < 8; ++i)
      Cs[(mt*16 + mo + i) * ldc + col] = acc[i] + bv;
  }
}

// ---------- utility kernels ----------
__global__ void k_cast(const float* __restrict__ s, __bf16* __restrict__ d, size_t n) {
  for (size_t i = blockIdx.x * (size_t)blockDim.x + threadIdx.x; i < n;
       i += (size_t)gridDim.x * blockDim.x)
    d[i] = (__bf16)s[i];
}

__global__ void k_transpose_cast(const float* __restrict__ s, __bf16* __restrict__ d,
                                 int R, int C) {  // s[R,C] -> d[C,R]
  int n = R * C;
  for (int i = blockIdx.x * blockDim.x + threadIdx.x; i < n; i += gridDim.x * blockDim.x) {
    int r = i / C, c = i % C;
    d[c * R + r] = (__bf16)s[i];
  }
}

__global__ void k_embed(const int* __restrict__ X, const float* __restrict__ emb,
                        __bf16* __restrict__ out, int rows) {
  size_t n = (size_t)rows * 256;
  for (size_t i = blockIdx.x * (size_t)blockDim.x + threadIdx.x; i < n;
       i += (size_t)gridDim.x * blockDim.x) {
    size_t r = i >> 8; int e = (int)(i & 255);
    out[i] = (__bf16)emb[(size_t)X[r] * 256 + e];
  }
}

__global__ void k_zero_int(int* p, int n) {
  int i = blockIdx.x * blockDim.x + threadIdx.x;
  if (i < n) p[i] = 0;
}

// ---------- generic WMMA GEMM: C[M,N] = A[M,K] @ W^T (+bias), W is [N,K] bf16 ----------
__global__ void k_gemm(const __bf16* __restrict__ A, const __bf16* __restrict__ W,
                       const float* __restrict__ bias, float* __restrict__ C,
                       int M, int N, int K, int ldc) {
  int lane = threadIdx.x & 31, wave = threadIdx.x >> 5;
  int ntiles = N >> 4;
  int total  = (M >> 4) * ntiles;
  int tile = blockIdx.x * 8 + wave;
  if (tile >= total) return;            // wave-uniform: EXEC stays all-ones for WMMA
  int mt = tile / ntiles, nt = tile % ntiles;
  f32x8 acc = zero8();
  const __bf16* wrow = W + (size_t)(nt*16 + (lane & 15)) * K;
  for (int k0 = 0; k0 < K; k0 += 32) {
    if (k0 + 32 < K) __builtin_prefetch(wrow + k0 + 32, 0, 1);
    acc = wmma_bf16(load_a_frag(A, K, mt*16, k0, lane),
                    load_b_frag(W, K, nt*16, k0, lane), acc);
  }
  int col = nt*16 + (lane & 15);
  int mo  = (lane & 16) ? 8 : 0;
  float bv = bias ? bias[col] : 0.f;
#pragma unroll
  for (int i = 0; i < 8; ++i)
    C[(size_t)(mt*16 + mo + i) * ldc + col] = acc[i] + bv;
}

// ---------- persistent GRU sequence (one block per direction) ----------
struct GruArgs {
  const float* gi0; const float* gi1;          // precomputed x@Wih^T + b_ih, [S,32,768]
  const __bf16* whh0; const __bf16* whh1;      // [768,256] bf16
  const float* bhh0; const float* bhh1;
  __bf16* ybf;                                 // [S,32,ycols], dir d -> cols d*256..
  float*  yf32;                                // same layout or nullptr
  __bf16* finals;                              // [32,512] slice, dir d -> cols d*256..
  int ycols; int S;
};

#define GRU_LDS (16384 + 32768 + 98304)        // hbf + hf + gh = 144 KB

__global__ void k_gru_seq(GruArgs ga) {
  extern __shared__ char smem[];
  __bf16* hbf = (__bf16*)smem;                 // [32,256] bf16
  float*  hf  = (float*)(smem + 16384);        // [32,256] f32
  float*  gh  = (float*)(smem + 16384 + 32768);// [32,768] f32
  int dir = blockIdx.x;
  const float*  gi  = dir ? ga.gi1  : ga.gi0;
  const __bf16* whh = dir ? ga.whh1 : ga.whh0;
  const float*  bhh = dir ? ga.bhh1 : ga.bhh0;
  int colOff = dir * 256;
  int tid = threadIdx.x, lane = tid & 31, wave = tid >> 5;

  for (int i = tid; i < 8192; i += 256) { hf[i] = 0.f; hbf[i] = (__bf16)0.f; }
  __syncthreads();

  for (int t = 0; t < ga.S; ++t) {
    int s = dir ? (ga.S - 1 - t) : t;
    // gh = h @ Whh^T + b_hh   (WMMA from LDS, 96 tiles over 8 waves)
    wave_gemm(hbf, whh, bhh, gh, 2, 48, 256, wave, lane);
    __syncthreads();
    const float* gis = gi + (size_t)s * (32 * 768);
    for (int idx = tid; idx < 8192; idx += 256) {
      int b = idx >> 8, j = idx & 255;
      const float* gib = gis + b * 768;
      const float* ghb = gh  + b * 768;
      float r  = sigm(gib[j]       + ghb[j]);
      float z  = sigm(gib[256 + j] + ghb[256 + j]);
      float nn = tanhf(gib[512 + j] + r * ghb[512 + j]);
      float h  = (1.f - z) * nn + z * hf[idx];
      hf[idx] = h;
      __bf16 hb = (__bf16)h;
      hbf[idx] = hb;
      size_t yo = ((size_t)s * 32 + b) * ga.ycols + colOff + j;
      ga.ybf[yo] = hb;
      if (ga.yf32) ga.yf32[yo] = h;
      if (t == ga.S - 1) ga.finals[b * 512 + colOff + j] = hb;
    }
    __syncthreads();
  }
}

// ---------- attention (one block per batch element) ----------
__global__ void k_attn(const float* __restrict__ hstate,   // [64,256]; rows 32..63 = h1
                       const float* __restrict__ Wl3, const float* __restrict__ bl3,
                       const float* __restrict__ va,
                       const float* __restrict__ w1e,      // [S,32,256]
                       const float* __restrict__ enc_out,  // [S,32,512]
                       __bf16* __restrict__ xcat) {        // [32,768]; write cols 256..767
  __shared__ float w2d[256];
  __shared__ float sc[400];
  __shared__ float red[256];
  int b = blockIdx.x, tid = threadIdx.x;
  const float* h1 = hstate + (32 + b) * 256;
  float acc = bl3[tid];
  const float* wr = Wl3 + tid * 256;
  for (int k = 0; k < 256; ++k) acc += wr[k] * h1[k];
  w2d[tid] = acc;
  __syncthreads();
  for (int s = tid; s < 400; s += 256) {
    const float* we = w1e + ((size_t)s * 32 + b) * 256;
    float sa = 0.f;
    for (int d = 0; d < 256; ++d) sa += va[d] * tanhf(we[d] + w2d[d]);
    sc[s] = sa;
  }
  __syncthreads();
  float m = -3.4e38f;
  for (int s = tid; s < 400; s += 256) m = fmaxf(m, sc[s]);
  red[tid] = m; __syncthreads();
  for (int o = 128; o > 0; o >>= 1) { if (tid < o) red[tid] = fmaxf(red[tid], red[tid+o]); __syncthreads(); }
  float gmax = red[0];
  __syncthreads();
  float ssum = 0.f;
  for (int s = tid; s < 400; s += 256) { float e = __expf(sc[s] - gmax); sc[s] = e; ssum += e; }
  red[tid] = ssum; __syncthreads();
  for (int o = 128; o > 0; o >>= 1) { if (tid < o) red[tid] += red[tid+o]; __syncthreads(); }
  float inv = 1.f / red[0];
  __syncthreads();
  for (int h = tid; h < 512; h += 256) {
    float x = 0.f;
    const float* eo = enc_out + (size_t)b * 512 + h;
    for (int s = 0; s < 400; ++s) x += sc[s] * eo[(size_t)s * 32 * 512];
    xcat[b * 768 + 256 + h] = (__bf16)(x * inv);
  }
}

// ---------- decoder cell (single block, all 5 GEMMs via WMMA in LDS) ----------
struct DecArgs {
  const int* dec_in;
  const __bf16* emb;        // dec_emb bf16 [V,256]
  __bf16* xcat;             // [32,768] global (cols 256..767 filled by k_attn)
  const __bf16* Wl2; const float* bl2;
  const __bf16* Wih0; const __bf16* Whh0; const float* bih0; const float* bhh0;
  const __bf16* Wih1; const __bf16* Whh1; const float* bih1; const float* bhh1;
  float* hstate;            // [64,256]
  __bf16* h1bf;             // [32,256] out, A-operand for logits GEMM
};

#define DEC_LDS 311296      // 304 KB of the 320 KB WGP LDS

__global__ void k_dec_cell(DecArgs da) {
  extern __shared__ char smem[];
  __bf16* h0b  = (__bf16*)smem;                 // 16 KB
  __bf16* h1b  = (__bf16*)(smem + 16384);       // 16 KB
  float*  h0f  = (float*)(smem + 32768);        // 32 KB
  float*  h1f  = (float*)(smem + 65536);        // 32 KB
  __bf16* xinb = (__bf16*)(smem + 98304);       // 16 KB
  float*  gi   = (float*)(smem + 114688);       // 96 KB
  float*  gh   = (float*)(smem + 212992);       // 96 KB
  int tid = threadIdx.x, lane = tid & 31, wave = tid >> 5;

  for (int i = tid; i < 8192; i += 256) {
    float a = da.hstate[i], c = da.hstate[8192 + i];
    h0f[i] = a; h0b[i] = (__bf16)a;
    h1f[i] = c; h1b[i] = (__bf16)c;
    int b = i >> 8, k = i & 255;
    da.xcat[b * 768 + k] = da.emb[(size_t)da.dec_in[b] * 256 + k];
  }
  __threadfence();
  __syncthreads();

  // xin = xcat @ Wl2^T + b_l2  (M=32,N=256,K=768) -> bf16 LDS
  for (int tt = wave; tt < 32; tt += 8) {
    int mt = tt >> 4, nt = tt & 15;
    f32x8 acc = zero8();
    for (int k0 = 0; k0 < 768; k0 += 32)
      acc = wmma_bf16(load_a_frag(da.xcat, 768, mt*16, k0, lane),
                      load_b_frag(da.Wl2, 768, nt*16, k0, lane), acc);
    int col = nt*16 + (lane & 15), mo = (lane & 16) ? 8 : 0;
    float bv = da.bl2[col];
#pragma unroll
    for (int i = 0; i < 8; ++i)
      xinb[(mt*16 + mo + i) * 256 + col] = (__bf16)(acc[i] + bv);
  }
  __syncthreads();

  // GRU cell layer 0
  wave_gemm(xinb, da.Wih0, da.bih0, gi, 2, 48, 256, wave, lane);
  wave_gemm(h0b,  da.Whh0, da.bhh0, gh, 2, 48, 256, wave, lane);
  __syncthreads();
  for (int idx = tid; idx < 8192; idx += 256) {
    int b = idx >> 8, j = idx & 255;
    const float* gib = gi + b * 768; const float* ghb = gh + b * 768;
    float r  = sigm(gib[j] + ghb[j]);
    float z  = sigm(gib[256 + j] + ghb[256 + j]);
    float nn = tanhf(gib[512 + j] + r * ghb[512 + j]);
    float h  = (1.f - z) * nn + z * h0f[idx];
    h0f[idx] = h; h0b[idx] = (__bf16)h;
  }
  __syncthreads();

  // GRU cell layer 1
  wave_gemm(h0b, da.Wih1, da.bih1, gi, 2, 48, 256, wave, lane);
  wave_gemm(h1b, da.Whh1, da.bhh1, gh, 2, 48, 256, wave, lane);
  __syncthreads();
  for (int idx = tid; idx < 8192; idx += 256) {
    int b = idx >> 8, j = idx & 255;
    const float* gib = gi + b * 768; const float* ghb = gh + b * 768;
    float r  = sigm(gib[j] + ghb[j]);
    float z  = sigm(gib[256 + j] + ghb[256 + j]);
    float nn = tanhf(gib[512 + j] + r * ghb[512 + j]);
    float h  = (1.f - z) * nn + z * h1f[idx];
    da.hstate[idx]        = h0f[idx];
    da.hstate[8192 + idx] = h;
    da.h1bf[idx]          = (__bf16)h;
  }
}

// ---------- argmax over 30000 per row ----------
__global__ void k_argmax(const float* __restrict__ logits, int* __restrict__ dec_in) {
  __shared__ float bv[256]; __shared__ int bi[256];
  int b = blockIdx.x, tid = threadIdx.x;
  const float* row = logits + (size_t)b * 30000;
  float best = -3.4e38f; int bidx = 0;
  for (int i = tid; i < 30000; i += 256) {
    float v = row[i];
    if (v > best) { best = v; bidx = i; }
  }
  bv[tid] = best; bi[tid] = bidx; __syncthreads();
  for (int o = 128; o > 0; o >>= 1) {
    if (tid < o) {
      if (bv[tid+o] > bv[tid] || (bv[tid+o] == bv[tid] && bi[tid+o] < bi[tid])) {
        bv[tid] = bv[tid+o]; bi[tid] = bi[tid+o];
      }
    }
    __syncthreads();
  }
  if (tid == 0) dec_in[b] = bi[0];
}

// ---------- host ----------
extern "C" void kernel_launch(void* const* d_in, const int* in_sizes, int n_in,
                              void* d_out, int out_size, void* d_ws, size_t ws_size,
                              hipStream_t stream) {
  (void)out_size; (void)ws_size;
  const int S = 400, B = 32, V = 30000;

  // Detect nested-dict flatten convention (insertion vs sorted keys) from sizes.
  bool sorted = (n_in > 10 && in_sizes[10] == 196608);
  int eW_ih[4], eW_hh[4], eB_ih[4], eB_hh[4];   // order: l0_f, l0_b, l1_f, l1_b
  {
    const int basesIns[4]  = {2, 6, 10, 14};
    const int basesSort[4] = {6, 2, 14, 10};    // sorted keys: l0_b,l0_f,l1_b,l1_f at 2,6,10,14
    for (int i = 0; i < 4; ++i) {
      if (!sorted) { int b = basesIns[i];  eW_ih[i]=b;   eW_hh[i]=b+1; eB_ih[i]=b+2; eB_hh[i]=b+3; }
      else         { int b = basesSort[i]; eW_hh[i]=b;   eW_ih[i]=b+1; eB_hh[i]=b+2; eB_ih[i]=b+3; }
    }
  }
  int dW_ih[2], dW_hh[2], dB_ih[2], dB_hh[2];
  for (int l = 0; l < 2; ++l) {
    int b = 20 + 4 * l;
    if (!sorted) { dW_ih[l]=b; dW_hh[l]=b+1; dB_ih[l]=b+2; dB_hh[l]=b+3; }
    else         { dW_hh[l]=b; dW_ih[l]=b+1; dB_hh[l]=b+2; dB_ih[l]=b+3; }
  }
  auto F = [&](int i) { return (const float*)d_in[i]; };
  const int*   X       = (const int*)d_in[0];
  const float* enc_emb = F(1);
  const float* W_eo    = F(18);
  const float* dec_emb = F(19);
  const float* b_out   = F(28);
  const float* Wm      = F(29);
  const float* v_attn  = F(30);
  const float* W_l2    = F(31);
  const float* b_l2    = F(32);
  const float* W_l3    = F(33);
  const float* b_l3    = F(34);

  // workspace allocator
  char* base = (char*)d_ws; size_t off = 0;
  auto alloc = [&](size_t bytes) -> void* {
    off = (off + 255) & ~(size_t)255;
    void* p = base + off; off += bytes; return p;
  };
  __bf16 *wih_bf[4], *whh_bf[4];
  for (int i = 0; i < 4; ++i) {
    wih_bf[i] = (__bf16*)alloc((size_t)in_sizes[eW_ih[i]] * 2);
    whh_bf[i] = (__bf16*)alloc((size_t)in_sizes[eW_hh[i]] * 2);
  }
  __bf16 *dwih_bf[2], *dwhh_bf[2];
  for (int l = 0; l < 2; ++l) {
    dwih_bf[l] = (__bf16*)alloc(196608 * 2);
    dwhh_bf[l] = (__bf16*)alloc(196608 * 2);
  }
  __bf16* Weo_bf = (__bf16*)alloc(131072 * 2);
  __bf16* WmT_bf = (__bf16*)alloc(131072 * 2);
  __bf16* Wl2_bf = (__bf16*)alloc(196608 * 2);
  __bf16* emb_bf = (__bf16*)alloc((size_t)V * 256 * 2);
  __bf16* x0_bf  = (__bf16*)alloc((size_t)S * B * 256 * 2);
  __bf16* y0_bf  = (__bf16*)alloc((size_t)S * B * 512 * 2);
  __bf16* y1_bf  = (__bf16*)alloc((size_t)S * B * 512 * 2);
  __bf16* fin_bf = (__bf16*)alloc(64 * 512 * 2);
  __bf16* xcat   = (__bf16*)alloc(32 * 768 * 2);
  __bf16* h1_bf  = (__bf16*)alloc(32 * 256 * 2);
  float* gi_f    = (float*)alloc((size_t)S * B * 768 * 4);
  float* gi_b    = (float*)alloc((size_t)S * B * 768 * 4);
  float* eo_f32  = (float*)alloc((size_t)S * B * 512 * 4);
  float* w1e     = (float*)alloc((size_t)S * B * 256 * 4);
  float* hstate  = (float*)alloc(64 * 256 * 4);
  int*   dec_in  = (int*)alloc(32 * 4);

  auto cast = [&](const float* s, __bf16* d, size_t n) {
    int blocks = (int)((n + 255) / 256); if (blocks > 4096) blocks = 4096;
    k_cast<<<blocks, 256, 0, stream>>>(s, d, n);
  };
  for (int i = 0; i < 4; ++i) {
    cast(F(eW_ih[i]), wih_bf[i], in_sizes[eW_ih[i]]);
    cast(F(eW_hh[i]), whh_bf[i], in_sizes[eW_hh[i]]);
  }
  for (int l = 0; l < 2; ++l) {
    cast(F(dW_ih[l]), dwih_bf[l], 196608);
    cast(F(dW_hh[l]), dwhh_bf[l], 196608);
  }
  cast(W_eo, Weo_bf, 131072);
  cast(W_l2, Wl2_bf, 196608);
  cast(dec_emb, emb_bf, (size_t)V * 256);
  k_transpose_cast<<<512, 256, 0, stream>>>(Wm, WmT_bf, 512, 256);  // [512,256]->[256,512]
  {
    size_t n = (size_t)S * B * 256;
    int blocks = (int)((n + 255) / 256); if (blocks > 8192) blocks = 8192;
    k_embed<<<blocks, 256, 0, stream>>>(X, enc_emb, x0_bf, S * B);
  }
  auto gemm = [&](const __bf16* A, const __bf16* W, const float* bias, float* C,
                  int M, int N, int K, int ldc) {
    int tiles = (M >> 4) * (N >> 4);
    int blocks = (tiles + 7) / 8;
    k_gemm<<<blocks, 256, 0, stream>>>(A, W, bias, C, M, N, K, ldc);
  };

  // encoder layer 0: input gates then recurrence (f & b concurrently, 2 blocks)
  gemm(x0_bf, wih_bf[0], F(eB_ih[0]), gi_f, S * B, 768, 256, 768);
  gemm(x0_bf, wih_bf[1], F(eB_ih[1]), gi_b, S * B, 768, 256, 768);
  {
    GruArgs ga;
    ga.gi0 = gi_f; ga.gi1 = gi_b;
    ga.whh0 = whh_bf[0]; ga.whh1 = whh_bf[1];
    ga.bhh0 = F(eB_hh[0]); ga.bhh1 = F(eB_hh[1]);
    ga.ybf = y0_bf; ga.yf32 = nullptr; ga.finals = fin_bf; ga.ycols = 512; ga.S = S;
    k_gru_seq<<<2, 256, GRU_LDS, stream>>>(ga);
  }
  // encoder layer 1 (reuse gi buffers)
  gemm(y0_bf, wih_bf[2], F(eB_ih[2]), gi_f, S * B, 768, 512, 768);
  gemm(y0_bf, wih_bf[3], F(eB_ih[3]), gi_b, S * B, 768, 512, 768);
  {
    GruArgs ga;
    ga.gi0 = gi_f; ga.gi1 = gi_b;
    ga.whh0 = whh_bf[2]; ga.whh1 = whh_bf[3];
    ga.bhh0 = F(eB_hh[2]); ga.bhh1 = F(eB_hh[3]);
    ga.ybf = y1_bf; ga.yf32 = eo_f32; ga.finals = fin_bf + 32 * 512; ga.ycols = 512; ga.S = S;
    k_gru_seq<<<2, 256, GRU_LDS, stream>>>(ga);
  }
  // attention precompute + decoder initial state
  gemm(y1_bf, WmT_bf, nullptr, w1e, S * B, 256, 512, 256);  // w1e = enc_out @ Wm
  gemm(fin_bf, Weo_bf, nullptr, hstate, 64, 256, 512, 256); // h_dec = finals @ W_eo^T
  k_zero_int<<<1, 32, 0, stream>>>(dec_in, 32);

  // decoder: 60 sequential steps (argmax feedback forces serialization)
  float* out = (float*)d_out;
  for (int t = 0; t < 60; ++t) {
    k_attn<<<32, 256, 0, stream>>>(hstate, W_l3, b_l3, v_attn, w1e, eo_f32, xcat);
    DecArgs da;
    da.dec_in = dec_in; da.emb = emb_bf; da.xcat = xcat;
    da.Wl2 = Wl2_bf; da.bl2 = b_l2;
    da.Wih0 = dwih_bf[0]; da.Whh0 = dwhh_bf[0]; da.bih0 = F(dB_ih[0]); da.bhh0 = F(dB_hh[0]);
    da.Wih1 = dwih_bf[1]; da.Whh1 = dwhh_bf[1]; da.bih1 = F(dB_ih[1]); da.bhh1 = F(dB_hh[1]);
    da.hstate = hstate; da.h1bf = h1_bf;
    k_dec_cell<<<1, 256, DEC_LDS, stream>>>(da);
    float* lt = out + (size_t)t * B * V;
    gemm(h1_bf, emb_bf, b_out, lt, 32, V, 256, V);   // tied-embedding logits, L2-resident B
    k_argmax<<<32, 256, 0, stream>>>(lt, dec_in);
  }
}
